// ParticleFlowNetwork_59485297049812
// MI455X (gfx1250) — compile-verified, matrix operand order checked by reading, not run on hardware
//
#include <hip/hip_runtime.h>
#include <stdint.h>

typedef __attribute__((ext_vector_type(16))) _Float16 v16h;
typedef __attribute__((ext_vector_type(8)))  float    v8f;

#define WMMA_F16(a, b, c) \
  __builtin_amdgcn_wmma_f32_16x16x32_f16(false, (a), false, (b), (short)0, (c), false, false)

// ---------------- workspace layout (element offsets) ----------------
// f16 weight table (transposed [N][K], zero padded)
#define OFF_W1T 0        // [128][32]
#define OFF_W2T 4096     // [128][128]
#define OFF_W3T 20480    // [128][128]
#define OFF_W4T 36864    // [64][128]
#define OFF_W5T 45056    // [256][64]
#define OFF_W6T 61440    // [256][256]
#define OFF_W7T 126976   // [128][256]
#define OFF_W8T 159744   // [128][128]
#define OFF_W9  176128   // [128]
#define WTAB_ELEMS 176256
#define WTAB_BYTES (WTAB_ELEMS * 2)   // 16B-aligned offsets throughout

// f32 bias table (padded)
#define OFF_B1 0      // 128
#define OFF_B2 128    // 128
#define OFF_B3 256    // 128
#define OFF_B4 384    // 64
#define OFF_B5 448    // 256
#define OFF_B6 704    // 256
#define OFF_B7 960    // 128
#define OFF_B8 1088   // 128
#define OFF_B9 1216   // 1
#define BTAB_FLOATS 1220
#define LATENT_BYTE_OFF (WTAB_BYTES + BTAB_FLOATS * 4)   // then 8192*64 f32

// ---------------- prep kernels ----------------
__global__ void pad_transpose_k(const float* __restrict__ src, _Float16* __restrict__ dst,
                                int K, int N, int Kp, int Np) {
  int i = blockIdx.x * blockDim.x + threadIdx.x;
  if (i >= Kp * Np) return;
  int n = i / Kp;
  int k = i - n * Kp;
  float v = (k < K && n < N) ? src[k * N + n] : 0.0f;
  dst[i] = (_Float16)v;
}

__global__ void pad_bias_k(const float* __restrict__ src, float* __restrict__ dst,
                           int N, int Np) {
  int i = blockIdx.x * blockDim.x + threadIdx.x;
  if (i >= Np) return;
  dst[i] = (i < N) ? src[i] : 0.0f;
}

// ---------------- async global->LDS staging (CDNA5 ASYNCcnt path) ----------------
__device__ __forceinline__ void async_ld16(void* lds_dst, const void* gsrc) {
  // LDS byte offset = low 32 bits of the flat shared-aperture address
  unsigned lo = (unsigned)(uintptr_t)lds_dst;
  unsigned long long ga = (unsigned long long)(uintptr_t)gsrc;
  asm volatile("global_load_async_to_lds_b128 %0, %1, off"
               :: "v"(lo), "v"(ga) : "memory");
}
__device__ __forceinline__ void async_wait0() {
  asm volatile("s_wait_asynccnt 0" ::: "memory");
}
__device__ __forceinline__ void copy_h_async(_Float16* dst, const _Float16* src, int n_elem) {
  int n16 = n_elem >> 3;                       // 8 halfs = 16B per transfer
  for (int i = threadIdx.x; i < n16; i += blockDim.x)
    async_ld16(dst + i * 8, src + i * 8);
}
__device__ __forceinline__ void copy_f(float* dst, const float* src, int n) {
  for (int i = threadIdx.x; i < n; i += blockDim.x) dst[i] = src[i];
}

// ---------------- WMMA fragment helpers ----------------
// 16x32 f16 fragment (A: rows=M; B stored transposed [N][K]: rows=N).
// lanes 0-15: K = {0..7, 16..23}; lanes 16-31: K = {8..15, 24..31}
__device__ __forceinline__ v16h load_frag(const _Float16* p, int ld) {
  int lane = threadIdx.x & 31;
  int r = lane & 15;
  int h = lane >> 4;
  const unsigned int* b = (const unsigned int*)(p + r * ld + h * 8);
  union { v16h v; unsigned int u[8]; } f;
  f.u[0] = b[0];  f.u[1] = b[1];  f.u[2] = b[2];  f.u[3] = b[3];
  f.u[4] = b[8];  f.u[5] = b[9];  f.u[6] = b[10]; f.u[7] = b[11];
  return f.v;
}

// D tile (16x16 f32): element (m = v + 8*(lane/16), n = lane%16)
__device__ __forceinline__ void store_relu(v8f acc, _Float16* dst, int ld, const float* bias) {
  int lane = threadIdx.x & 31;
  int n = lane & 15;
  int h = lane >> 4;
  float bb = bias[n];
  #pragma unroll
  for (int v = 0; v < 8; ++v) {
    float t = acc[v] + bb;
    t = t > 0.0f ? t : 0.0f;
    dst[(v + 8 * h) * ld + n] = (_Float16)t;
  }
}

// Ping-pong ReLU layer: A fragments hoisted to registers, out-of-place store.
template <int NT, int KS>
__device__ __forceinline__ void layer_pp(const _Float16* Zin, int ldin,
                                         _Float16* Zout, int ldout,
                                         const _Float16* W, const float* bias, int rows) {
  v16h a[KS];
  #pragma unroll
  for (int ks = 0; ks < KS; ++ks) a[ks] = load_frag(Zin + rows * ldin + ks * 32, ldin);
  #pragma unroll
  for (int nt = 0; nt < NT; ++nt) {
    v8f acc = {};
    #pragma unroll
    for (int ks = 0; ks < KS; ++ks) {
      v16h b = load_frag(W + (nt * 16) * (KS * 32) + ks * 32, KS * 32);
      acc = WMMA_F16(a[ks], b, acc);
    }
    store_relu(acc, Zout + rows * ldout + nt * 16, ldout, bias + nt * 16);
  }
}

// In-place ReLU layer (square): stash the wave's 16-row output in VGPRs,
// write back over the input rows after all tiles are computed.
template <int NT, int KS>
__device__ __forceinline__ void layer_ip(_Float16* Z, int ld,
                                         const _Float16* W, const float* bias, int rows) {
  v16h a[KS];
  #pragma unroll
  for (int ks = 0; ks < KS; ++ks) a[ks] = load_frag(Z + rows * ld + ks * 32, ld);
  int lane = threadIdx.x & 31, n = lane & 15, h = lane >> 4;
  _Float16 st[NT * 8];
  #pragma unroll
  for (int nt = 0; nt < NT; ++nt) {
    v8f acc = {};
    #pragma unroll
    for (int ks = 0; ks < KS; ++ks) {
      v16h b = load_frag(W + (nt * 16) * (KS * 32) + ks * 32, KS * 32);
      acc = WMMA_F16(a[ks], b, acc);
    }
    float bb = bias[nt * 16 + n];
    #pragma unroll
    for (int v = 0; v < 8; ++v) {
      float t = acc[v] + bb;
      t = t > 0.0f ? t : 0.0f;
      st[nt * 8 + v] = (_Float16)t;
    }
  }
  #pragma unroll
  for (int nt = 0; nt < NT; ++nt) {
    #pragma unroll
    for (int v = 0; v < 8; ++v)
      Z[(rows + v + 8 * h) * ld + nt * 16 + n] = st[nt * 8 + v];
  }
}

// ---------------- kernel 1: per-particle MLP + masked sigmoid-sum ----------------
// 1 block = 1 event (128 particles), 8 waves x 16 rows. ~130KB LDS -> 2 blocks/WGP.
__global__ void __launch_bounds__(256) pfn_particle(const float* __restrict__ x,
                                                    const _Float16* __restrict__ wtab,
                                                    const float* __restrict__ btab,
                                                    float* __restrict__ latent) {
  __shared__ __align__(16) _Float16 sW1[128 * 32];
  __shared__ __align__(16) _Float16 sW2[128 * 128];
  __shared__ __align__(16) _Float16 sW3[128 * 128];
  __shared__ __align__(16) _Float16 sW4[64 * 128];
  __shared__ __align__(16) _Float16 sX[128 * 32];
  __shared__ __align__(16) _Float16 sH[128 * 128];
  __shared__ float sB1[128], sB2[128], sB3[128], sB4[64];
  __shared__ float sMask[128];
  __shared__ float sLat[64];

  copy_h_async(sW1, wtab + OFF_W1T, 128 * 32);
  copy_h_async(sW2, wtab + OFF_W2T, 128 * 128);
  copy_h_async(sW3, wtab + OFF_W3T, 128 * 128);
  copy_h_async(sW4, wtab + OFF_W4T, 64 * 128);
  copy_f(sB1, btab + OFF_B1, 128);
  copy_f(sB2, btab + OFF_B2, 128);
  copy_f(sB3, btab + OFF_B3, 128);
  copy_f(sB4, btab + OFF_B4, 64);

  const float* xe = x + (size_t)blockIdx.x * 512;   // [128 particles][4 feats]
  for (int i = threadIdx.x; i < 128 * 32; i += 256) {
    int p = i >> 5, k = i & 31;
    float v = 0.0f;
    if (k < 4) {
      v = xe[p * 4 + k];
      if (!(v == v)) v = 0.0f;   // NaN -> 0
    }
    if (k == 0) {
      float v0 = xe[p * 4];
      sMask[p] = (v0 == v0) ? 1.0f : 0.0f;
    }
    sX[p * 32 + k] = (_Float16)v;
  }
  if (threadIdx.x < 64) sLat[threadIdx.x] = 0.0f;
  async_wait0();
  __syncthreads();

  int rows = (threadIdx.x >> 5) * 16;   // wave's 16 particles
  layer_pp<8, 1>(sX, 32, sH, 128, sW1, sB1, rows);   // 4(->32) -> 100(->128)
  layer_ip<8, 4>(sH, 128, sW2, sB2, rows);           // 100 -> 100 (in place)
  layer_ip<8, 4>(sH, 128, sW3, sB3, rows);           // 100 -> 100 (in place)

  // layer 4: 100 -> 64, sigmoid, masked sum over particles
  int lane = threadIdx.x & 31;
  int ncol = lane & 15, h = lane >> 4;
  v16h a4[4];
  #pragma unroll
  for (int ks = 0; ks < 4; ++ks) a4[ks] = load_frag(sH + rows * 128 + ks * 32, 128);
  #pragma unroll
  for (int nt = 0; nt < 4; ++nt) {
    v8f acc = {};
    #pragma unroll
    for (int ks = 0; ks < 4; ++ks) {
      v16h b = load_frag(sW4 + (nt * 16) * 128 + ks * 32, 128);
      acc = WMMA_F16(a4[ks], b, acc);
    }
    float bb = sB4[nt * 16 + ncol];
    float ps = 0.0f;
    #pragma unroll
    for (int v = 0; v < 8; ++v) {
      int m = rows + v + 8 * h;
      float t = 1.0f / (1.0f + __expf(-(acc[v] + bb)));
      ps += sMask[m] * t;
    }
    atomicAdd(&sLat[nt * 16 + ncol], ps);
  }
  __syncthreads();
  if (threadIdx.x < 64)
    latent[(size_t)blockIdx.x * 64 + threadIdx.x] = sLat[threadIdx.x];
}

// ---------------- kernel 2: classifier 64->256->256->128->128->1 ----------------
__global__ void __launch_bounds__(256) pfn_classifier(const float* __restrict__ latent,
                                                      const _Float16* __restrict__ wtab,
                                                      const float* __restrict__ btab,
                                                      float* __restrict__ out) {
  __shared__ __align__(16) _Float16 sW[256 * 256 + 128];  // largest layer + W9
  __shared__ __align__(16) _Float16 sZa[128 * 256];
  __shared__ __align__(16) _Float16 sZb[128 * 256];
  __shared__ float sB[260];

  int r0 = blockIdx.x * 128;
  for (int i = threadIdx.x; i < 128 * 64; i += 256)
    sZa[i] = (_Float16)latent[(size_t)r0 * 64 + i];

  copy_h_async(sW, wtab + OFF_W5T, 256 * 64);
  copy_f(sB, btab + OFF_B5, 256);
  async_wait0();
  __syncthreads();

  int rows = (threadIdx.x >> 5) * 16;
  layer_pp<16, 2>(sZa, 64, sZb, 256, sW, sB, rows);
  __syncthreads();

  copy_h_async(sW, wtab + OFF_W6T, 256 * 256);
  copy_f(sB, btab + OFF_B6, 256);
  async_wait0();
  __syncthreads();
  layer_pp<16, 8>(sZb, 256, sZa, 256, sW, sB, rows);
  __syncthreads();

  copy_h_async(sW, wtab + OFF_W7T, 128 * 256);
  copy_f(sB, btab + OFF_B7, 128);
  async_wait0();
  __syncthreads();
  layer_pp<8, 8>(sZa, 256, sZb, 128, sW, sB, rows);
  __syncthreads();

  copy_h_async(sW, wtab + OFF_W8T, 128 * 128);
  copy_h_async(sW + 128 * 128, wtab + OFF_W9, 128);
  copy_f(sB, btab + OFF_B8, 128);
  copy_f(sB + 128, btab + OFF_B9, 1);
  async_wait0();
  __syncthreads();
  layer_pp<8, 4>(sZb, 128, sZa, 128, sW, sB, rows);
  __syncthreads();

  if (threadIdx.x < 128) {
    int r = threadIdx.x;
    float acc = sB[128];                       // b9
    const _Float16* z = sZa + r * 128;
    const _Float16* w9 = sW + 128 * 128;
    for (int k = 0; k < 128; ++k) acc += (float)z[k] * (float)w9[k];
    out[r0 + r] = 1.0f / (1.0f + __expf(-acc));
  }
}

// ---------------- host launcher ----------------
extern "C" void kernel_launch(void* const* d_in, const int* in_sizes, int n_in,
                              void* d_out, int out_size, void* d_ws, size_t ws_size,
                              hipStream_t stream) {
  (void)in_sizes; (void)n_in; (void)out_size; (void)ws_size;
  const float* x = (const float*)d_in[0];
  const float* W[9];
  const float* b[9];
  for (int i = 0; i < 9; ++i) {
    W[i] = (const float*)d_in[1 + 2 * i];
    b[i] = (const float*)d_in[2 + 2 * i];
  }
  _Float16* wtab = (_Float16*)d_ws;
  float* btab = (float*)((char*)d_ws + WTAB_BYTES);
  float* latent = (float*)((char*)d_ws + LATENT_BYTE_OFF);

  struct WCfg { int K, N, Kp, Np, off; };
  const WCfg wc[9] = {
    {4,   100, 32,  128, OFF_W1T},
    {100, 100, 128, 128, OFF_W2T},
    {100, 100, 128, 128, OFF_W3T},
    {100, 64,  128, 64,  OFF_W4T},
    {64,  256, 64,  256, OFF_W5T},
    {256, 256, 256, 256, OFF_W6T},
    {256, 128, 256, 128, OFF_W7T},
    {128, 128, 128, 128, OFF_W8T},
    {128, 1,   128, 1,   OFF_W9},
  };
  struct BCfg { int N, Np, off; };
  const BCfg bc[9] = {
    {100, 128, OFF_B1}, {100, 128, OFF_B2}, {100, 128, OFF_B3}, {64, 64, OFF_B4},
    {256, 256, OFF_B5}, {256, 256, OFF_B6}, {128, 128, OFF_B7}, {128, 128, OFF_B8},
    {1, 1, OFF_B9},
  };
  for (int i = 0; i < 9; ++i) {
    int tot = wc[i].Kp * wc[i].Np;
    pad_transpose_k<<<(tot + 255) / 256, 256, 0, stream>>>(
        W[i], wtab + wc[i].off, wc[i].K, wc[i].N, wc[i].Kp, wc[i].Np);
    pad_bias_k<<<(bc[i].Np + 255) / 256, 256, 0, stream>>>(
        b[i], btab + bc[i].off, bc[i].N, bc[i].Np);
  }
  pfn_particle<<<8192, 256, 0, stream>>>(x, wtab, btab, latent);
  pfn_classifier<<<64, 256, 0, stream>>>(latent, wtab, btab, (float*)d_out);
}
